// TDANetBlock_50319836840337
// MI455X (gfx1250) — matrix operand order repaired and checked by reading, not compiled, not run on hardware
//
#include <hip/hip_runtime.h>
#include <hip/hip_bf16.h>
#include <math.h>

// ---------------------------------------------------------------------------
// Types for CDNA5 WMMA (wave32): bf16 A/B, f32 accumulate. TDM descriptors.
// ---------------------------------------------------------------------------
typedef __bf16 bf16_t;
typedef bf16_t v16bf __attribute__((ext_vector_type(16)));
typedef float  v8f   __attribute__((ext_vector_type(8)));
typedef unsigned int u32;
typedef u32 v4u __attribute__((ext_vector_type(4)));
typedef int v4i __attribute__((ext_vector_type(4)));
typedef int v8i __attribute__((ext_vector_type(8)));

__device__ __forceinline__ float gelu_f(float x) {
    return 0.5f * x * (1.0f + erff(x * 0.70710678118654752f));
}
__device__ __forceinline__ float sigmoid_f(float x) {
    return 1.0f / (1.0f + expf(-x));
}

// ---------------------------------------------------------------------------
// WMMA GEMM: Y[b,m,n] = act( sum_k W[m,k]*X[b,k,n] + bias[m] ) (+R[b,m,n])
// One wave computes a 16(M) x 64(N) tile: 4 WMMAs per K-chunk of 32.
// B tiles (32 x 64 f32) are staged into LDS by the Tensor Data Mover with
// 1-DWORD row padding (rows at stride 65 floats -> no LDS bank conflicts).
// M%16==0, N%64==0, K%32==0 (true for all call sites here).
// ---------------------------------------------------------------------------
__global__ void gemm_wmma_kernel(const float* __restrict__ W,
                                 const float* __restrict__ bias,
                                 const float* __restrict__ X,
                                 float* __restrict__ Y,
                                 int M, int K, int N, int act,
                                 const float* __restrict__ R) {
    const int lane   = threadIdx.x;          // 0..31
    const int tilesN = N >> 6;               // 64-wide N tiles
    const int tilesM = M >> 4;
    const int bid = blockIdx.x;
    const int tn = bid % tilesN;
    const int tm = (bid / tilesN) % tilesM;
    const int b  = bid / (tilesN * tilesM);
    const int m0 = tm << 4, n0 = tn << 6;

    __shared__ float tb[32][65];             // K-chunk x 64 cols, +1 pad (TDM pad)

    const float* __restrict__ Xb = X + (size_t)b * K * N;

    // ---- Tensor DMA descriptor (D#), per CDNA5 ISA ----
    const u32 ldsOff = (u32)(unsigned long long)(&tb[0][0]);   // low 32b = LDS offset
    const unsigned long long gbase =
        (unsigned long long)(const char*)(Xb + n0);

    v8i g1 = {};
    // [15:0] wg_mask=0 | [17:16] data_size=2 (4B) | [20] pad_enable
    // [24:22] pad_interval=5 (64 DWORDs) | [31:25] pad_amount=0 (1 DWORD)
    g1[0] = (int)((2u << 16) | (1u << 20) | (5u << 22));
    g1[1] = (int)(((u32)N & 0xFFFFu) << 16);                 // tensor_dim0[15:0]
    g1[2] = (int)(((u32)N >> 16) | (((u32)K & 0xFFFFu) << 16)); // dim0[31:16]|dim1[15:0]
    g1[3] = (int)(((u32)K >> 16) | (64u << 16));             // dim1[31:16]|tile_dim0=64
    g1[4] = 32;                                              // tile_dim1=32, tile_dim2=0
    g1[5] = (int)(u32)N;                                     // tensor_dim0_stride[31:0]
    g1[6] = 0;                                               // stride0[47:32]|stride1[15:0]
    g1[7] = 0;
    const v4i gz4 = {};
    const v8i gz8 = {};
    (void)gz8;

    v8f acc[4] = {};
    const int half = lane >> 4;              // 0: lanes 0-15, 1: lanes 16-31
    const int lm   = lane & 15;
    const int kbA  = half ? 8 : 0;           // A K sub-offset per ISA layout
    const int kbB  = half ? 16 : 0;          // unused; B reads below index rows with kbA
    const float* __restrict__ Wrow = W + (size_t)(m0 + lm) * K;

    for (int kk = 0; kk < K; kk += 32) {
        // ---- TDM: async load 32x64 f32 tile (rows kk..kk+31) into LDS ----
        const unsigned long long ga = gbase + (unsigned long long)kk * (u32)N * 4ull;
        v4u g0;
        g0[0] = 1u;                          // count=1 (valid user descriptor)
        g0[1] = ldsOff;                      // lds_addr
        g0[2] = (u32)(ga & 0xFFFFFFFFull);   // global_addr[31:0]
        g0[3] = (u32)(ga >> 32) | (2u << 30);// global_addr[56:32] | type=2
#if defined(__clang_major__) && __clang_major__ >= 23
        __builtin_amdgcn_tensor_load_to_lds(g0, g1, gz4, gz4, gz8, 0);
#else
        __builtin_amdgcn_tensor_load_to_lds(g0, g1, gz4, gz4, 0);
#endif

        // ---- A (16x32 bf16) direct from global while TDM is in flight ----
        // lanes0-15: K=kk+{0..7,16..23}; lanes16-31: K=kk+{8..15,24..31}
        v16bf a;
        #pragma unroll
        for (int i = 0; i < 8; ++i) {
            a[i]     = (bf16_t)Wrow[kk + kbA + i];
            a[i + 8] = (bf16_t)Wrow[kk + 16 + kbA + i];
        }
        if (kk + 32 < K) {                   // gfx1250 prefetch: next A chunk
            __builtin_prefetch(Wrow + kk + 32, 0, 1);
        }

        __builtin_amdgcn_s_wait_tensorcnt(0);        // TDM done writing LDS
        asm volatile("" ::: "memory");               // keep LDS reads below

        // ---- B (32x16 bf16) x4 from LDS; lane n = column per ISA layout ----
        #pragma unroll
        for (int j = 0; j < 4; ++j) {
            v16bf bv;
            #pragma unroll
            for (int i = 0; i < 16; ++i) bv[i] = (bf16_t)tb[(i < 8) ? (kbA + i) : (16 + kbA + (i - 8))][lm + 16 * j];
            acc[j] = __builtin_amdgcn_wmma_f32_16x16x32_bf16(
                         false, a, false, bv, (short)0, acc[j], false, false);
        }

        // LDS reads must land before next TDM overwrites the tile
        asm volatile("s_wait_dscnt 0x0" ::: "memory");
    }

    // D layout: VGPR r -> M = m0 + r (+8 for upper half-wave), N = col
    #pragma unroll
    for (int r = 0; r < 8; ++r) {
        const int row = m0 + r + (half ? 8 : 0);
        const float bs = bias[row];
        #pragma unroll
        for (int j = 0; j < 4; ++j) {
            const int col = n0 + lm + 16 * j;
            float v = acc[j][r] + bs;
            if (act == 1) v = gelu_f(v);
            const size_t oi = ((size_t)b * M + row) * N + col;
            if (R) v += R[oi];
            Y[oi] = v;
        }
    }
}

// ---------------------------------------------------------------------------
// Per-batch mean / variance over [C*T] (for GlobLN). grid = B, block = 256.
// ---------------------------------------------------------------------------
__global__ void stats_kernel(const float* __restrict__ x, float* __restrict__ st,
                             int perBatch) {
    const int b = blockIdx.x;
    const float* p = x + (size_t)b * perBatch;
    float s = 0.f, s2 = 0.f;
    for (int i = threadIdx.x; i < perBatch; i += blockDim.x) {
        float v = p[i]; s += v; s2 += v * v;
    }
    __shared__ float sh[256], sh2[256];
    int t = threadIdx.x;
    sh[t] = s; sh2[t] = s2; __syncthreads();
    for (int o = 128; o > 0; o >>= 1) {
        if (t < o) { sh[t] += sh[t + o]; sh2[t] += sh2[t + o]; }
        __syncthreads();
    }
    if (t == 0) {
        float m = sh[0] / perBatch;
        st[2 * b] = m;
        st[2 * b + 1] = sh2[0] / perBatch - m * m;
    }
}

// GlobLN apply (+ optional PReLU), elementwise, may run in place.
__global__ void globln_kernel(const float* __restrict__ x, const float* __restrict__ st,
                              const float* __restrict__ g, const float* __restrict__ bt,
                              const float* __restrict__ pa, float* __restrict__ y,
                              int C, int T, int total) {
    int i = blockIdx.x * blockDim.x + threadIdx.x;
    if (i >= total) return;
    int c = (i / T) % C;
    int b = i / (C * T);
    float v = (x[i] - st[2 * b]) * rsqrtf(st[2 * b + 1] + 1e-8f);
    v = g[c] * v + bt[c];
    if (pa) { float a = pa[0]; v = (v >= 0.f) ? v : a * v; }
    y[i] = v;
}

// Depthwise conv1d (groups == C), optional bias.
__global__ void dwconv_kernel(const float* __restrict__ x, const float* __restrict__ w,
                              const float* __restrict__ bias, float* __restrict__ y,
                              int C, int Tin, int Tout, int Kw, int stride, int pad,
                              int total) {
    int i = blockIdx.x * blockDim.x + threadIdx.x;
    if (i >= total) return;
    int t = i % Tout;
    int c = (i / Tout) % C;
    int b = i / (C * Tout);
    const float* xr = x + ((size_t)b * C + c) * Tin;
    float acc = bias ? bias[c] : 0.f;
    int base = t * stride - pad;
    for (int j = 0; j < Kw; ++j) {
        int s = base + j;
        if (s >= 0 && s < Tin) acc += w[c * Kw + j] * xr[s];
    }
    y[i] = acc;
}

// Adaptive avg-pool (block mean) down to Lg, accumulate or write.
__global__ void pool_kernel(const float* __restrict__ f, float* __restrict__ gf,
                            int Tin, int Lg, int accum, int total) {
    int i = blockIdx.x * blockDim.x + threadIdx.x;
    if (i >= total) return;
    int l = i % Lg;
    int bc = i / Lg;
    int w = Tin / Lg;
    const float* p = f + (size_t)bc * Tin + (size_t)l * w;
    float s = 0.f;
    for (int j = 0; j < w; ++j) s += p[j];
    s /= (float)w;
    gf[i] = accum ? gf[i] + s : s;
}

// LayerNorm over channel dim for [B,C,T]; grid = B*T, block = 256.
__global__ void lnch_kernel(const float* __restrict__ x, const float* __restrict__ w,
                            const float* __restrict__ bvec, float* __restrict__ y,
                            int C, int T) {
    const int bt = blockIdx.x;
    const int b = bt / T, t = bt % T;
    const float* xp = x + (size_t)b * C * T + t;
    float s = 0.f, s2 = 0.f;
    for (int c = threadIdx.x; c < C; c += blockDim.x) {
        float v = xp[(size_t)c * T]; s += v; s2 += v * v;
    }
    __shared__ float sh[256], sh2[256];
    __shared__ float mv[2];
    int tid = threadIdx.x;
    sh[tid] = s; sh2[tid] = s2; __syncthreads();
    for (int o = 128; o > 0; o >>= 1) {
        if (tid < o) { sh[tid] += sh[tid + o]; sh2[tid] += sh2[tid + o]; }
        __syncthreads();
    }
    if (tid == 0) {
        float m = sh[0] / C;
        mv[0] = m;
        mv[1] = rsqrtf(sh2[0] / C - m * m + 1e-5f);
    }
    __syncthreads();
    float* yp = y + (size_t)b * C * T + t;
    for (int c = threadIdx.x; c < C; c += blockDim.x) {
        float v = xp[(size_t)c * T];
        yp[(size_t)c * T] = w[c] * (v - mv[0]) * mv[1] + bvec[c];
    }
}

// S4D kernel construction: kf/kb[h,l] = 2*Re( sum_n C*(exp(dtA)-1)/A * exp(dtA*l) )
__global__ void s4kern_kernel(const float* __restrict__ log_dt,
                              const float* __restrict__ log_A_real,
                              const float* __restrict__ A_imag,
                              const float* __restrict__ C_re,
                              const float* __restrict__ C_im,
                              float* __restrict__ kf, float* __restrict__ kb,
                              int H, int Nst, int L) {
    int i = blockIdx.x * blockDim.x + threadIdx.x;
    if (i >= H * L) return;
    int h = i / L, l = i % L;
    float dt = expf(log_dt[h]);
    float af = 0.f, abk = 0.f;
    for (int n = 0; n < Nst; ++n) {
        float Ar = -expf(log_A_real[h * Nst + n]);
        float Ai = A_imag[h * Nst + n];
        float dAr = Ar * dt, dAi = Ai * dt;
        float er = expf(dAr);
        float ec = er * cosf(dAi), es = er * sinf(dAi);
        float den = Ar * Ar + Ai * Ai;
        float br = ((ec - 1.f) * Ar + es * Ai) / den;     // Bd = (e^{dtA}-1)/A
        float bi = (es * Ar - (ec - 1.f) * Ai) / den;
        float El = expf(dAr * (float)l);
        float Ec = El * cosf(dAi * (float)l), Es = El * sinf(dAi * (float)l);
        float cr = C_re[((size_t)0 * H + h) * Nst + n];
        float ci = C_im[((size_t)0 * H + h) * Nst + n];
        float pr = cr * br - ci * bi, pi = cr * bi + ci * br;
        af += pr * Ec - pi * Es;
        cr = C_re[((size_t)1 * H + h) * Nst + n];
        ci = C_im[((size_t)1 * H + h) * Nst + n];
        pr = cr * br - ci * bi; pi = cr * bi + ci * br;
        abk += pr * Ec - pi * Es;
    }
    kf[(size_t)h * L + l] = 2.f * af;
    kb[(size_t)h * L + l] = 2.f * abk;
}

// Bidirectional SSM conv (equivalent of the FFT circular conv) + D skip + GELU.
// grid = B*H, block = L (=256). LDS tiles of u, kf, kb.
__global__ void s4conv_kernel(const float* __restrict__ u,
                              const float* __restrict__ kf,
                              const float* __restrict__ kb,
                              const float* __restrict__ D,
                              float* __restrict__ y, int H, int L) {
    const int bh = blockIdx.x;
    const int h = bh % H;
    __shared__ float su[256], skf[256], skb[256];
    const int t = threadIdx.x;
    const float* ur = u + (size_t)bh * L;
    su[t]  = ur[t];
    skf[t] = kf[(size_t)h * L + t];
    skb[t] = kb[(size_t)h * L + t];
    __syncthreads();
    float acc = 0.f;
    for (int s = 0; s < L; ++s)
        acc += su[s] * ((s <= t) ? skf[t - s] : skb[s - t - 1]);
    y[(size_t)bh * L + t] = gelu_f(acc + D[h] * su[t]);
}

// GLU over channel split of [B,2H,L] -> [B,H,L]
__global__ void glu_kernel(const float* __restrict__ in, float* __restrict__ out,
                           int H, int L, int total) {
    int i = blockIdx.x * blockDim.x + threadIdx.x;
    if (i >= total) return;
    int l = i % L;
    int h = (i / L) % H;
    int b = i / (H * L);
    float a = in[((size_t)b * 2 * H + h) * L + l];
    float g = in[((size_t)b * 2 * H + H + h) * L + l];
    out[i] = a * sigmoid_f(g);
}

__global__ void add_kernel(float* __restrict__ y, const float* __restrict__ x, int total) {
    int i = blockIdx.x * blockDim.x + threadIdx.x;
    if (i < total) y[i] += x[i];
}

// LA combine: out = LN(local) * sigmoid(LN(gact))[interp] + LN(gfeat)[interp]
__global__ void la_combine_kernel(const float* __restrict__ L, const float* __restrict__ sL,
                                  const float* __restrict__ lg, const float* __restrict__ lb,
                                  const float* __restrict__ A_, const float* __restrict__ sA,
                                  const float* __restrict__ ag, const float* __restrict__ ab,
                                  const float* __restrict__ G, const float* __restrict__ sG,
                                  const float* __restrict__ gg, const float* __restrict__ gb,
                                  float* __restrict__ out, int C, int Tl, int Tg, int total) {
    int i = blockIdx.x * blockDim.x + threadIdx.x;
    if (i >= total) return;
    int t = i % Tl;
    int c = (i / Tl) % C;
    int b = i / (C * Tl);
    float lv = lg[c] * ((L[i] - sL[2 * b]) * rsqrtf(sL[2 * b + 1] + 1e-8f)) + lb[c];
    int tg = (int)(((long long)t * Tg) / Tl);     // nearest: floor(t*S/T)
    size_t ig = ((size_t)b * C + c) * Tg + tg;
    float av = ag[c] * ((A_[ig] - sA[2 * b]) * rsqrtf(sA[2 * b + 1] + 1e-8f)) + ab[c];
    float gv = gg[c] * ((G[ig] - sG[2 * b]) * rsqrtf(sG[2 * b + 1] + 1e-8f)) + gb[c];
    out[i] = lv * sigmoid_f(av) + gv;
}

// ---------------------------------------------------------------------------
// Host orchestration
// ---------------------------------------------------------------------------
extern "C" void kernel_launch(void* const* d_in, const int* in_sizes, int n_in,
                              void* d_out, int out_size, void* d_ws, size_t ws_size,
                              hipStream_t stream) {
    (void)in_sizes; (void)n_in; (void)out_size; (void)ws_size;
    const int Bn = 8, CIN = 128, CH = 512, Tt = 2048, Lg = 256, NST = 16;

    // Input indices (setup_inputs() recursive insertion order)
    const int I_X = 0, I_PW = 1, I_PB = 2, I_PG = 3, I_PBETA = 4, I_PA = 5;
    auto SPP  = [&](int k, int j) { return 6 + 4 * k + j; };   // w,b,g,beta
    auto FUSP = [&](int i) { return 22 + 9 * i; };             // lw,lg,lb,aw,ag,ab,gw,gg,gb
    auto LSTP = [&](int i) { return 58 + 9 * i; };
    const int I_S4 = 85;  // ln_w,ln_b,log_dt,log_A_real,A_imag,C_re,C_im,D,ow,ob
    const int I_FF = 95;  // ln_w,ln_b,w1,b1,w2,b2
    const int I_RW = 101, I_RB = 102;
    auto P = [&](int i) { return (const float*)d_in[i]; };

    // Bump allocator over workspace
    char* wsb = (char*)d_ws; size_t off = 0;
    auto alloc = [&](size_t n) {
        float* p = (float*)(wsb + off);
        off += ((n * sizeof(float) + 255) & ~(size_t)255);
        return p;
    };
    float* o    = alloc((size_t)Bn * CH * Tt);
    float* f0   = alloc((size_t)Bn * CH * Tt);
    float* f1   = alloc((size_t)Bn * CH * 1024);
    float* f2   = alloc((size_t)Bn * CH * 512);
    float* f3   = alloc((size_t)Bn * CH * 256);
    float* gf   = alloc((size_t)Bn * CH * 256);
    float* t1   = alloc((size_t)Bn * CH * 256);
    float* t2   = alloc((size_t)Bn * 2 * CH * 256);
    float* t3   = alloc((size_t)Bn * CH * 256);
    float* kf   = alloc((size_t)CH * 256);
    float* kb   = alloc((size_t)CH * 256);
    float* fu0  = alloc((size_t)Bn * CH * Tt);
    float* fu1  = alloc((size_t)Bn * CH * 1024);
    float* fu2  = alloc((size_t)Bn * CH * 512);
    float* fu3  = alloc((size_t)Bn * CH * 256);
    float* e2   = alloc((size_t)Bn * CH * 512);
    float* e1   = alloc((size_t)Bn * CH * 1024);
    float* e0   = alloc((size_t)Bn * CH * Tt);
    float* tmpL = alloc((size_t)Bn * CH * Tt);
    float* tmpA = alloc((size_t)Bn * CH * 1024);
    float* tmpG = alloc((size_t)Bn * CH * 1024);
    float* st0  = alloc(2 * Bn);
    float* st1  = alloc(2 * Bn);
    float* st2  = alloc(2 * Bn);

    auto blocks = [&](int total) { return dim3((total + 255) / 256); };

    auto gemm = [&](const float* W, const float* bias, const float* X, float* Y,
                    int M, int K, int N, int act, const float* R) {
        dim3 grid((N / 64) * (M / 16) * Bn);
        gemm_wmma_kernel<<<grid, 32, 0, stream>>>(W, bias, X, Y, M, K, N, act, R);
    };
    auto stats = [&](const float* x, float* st, int perBatch) {
        stats_kernel<<<Bn, 256, 0, stream>>>(x, st, perBatch);
    };
    auto globln = [&](const float* x, float* st, const float* g, const float* bt,
                      const float* pa, float* y, int T) {
        int tot = Bn * CH * T;
        globln_kernel<<<blocks(tot), 256, 0, stream>>>(x, st, g, bt, pa, y, CH, T, tot);
    };
    auto dwconv = [&](const float* x, const float* w, const float* b, float* y,
                      int Tin, int Tout, int Kw, int stride, int pad) {
        int tot = Bn * CH * Tout;
        dwconv_kernel<<<blocks(tot), 256, 0, stream>>>(x, w, b, y, CH, Tin, Tout,
                                                       Kw, stride, pad, tot);
    };
    auto la = [&](const float* xl, int Tl, const float* xg, int Tg, int pb, int Kw,
                  float* out) {
        int pad = (Kw - 1) / 2;
        dwconv(xl, P(pb + 0), nullptr, tmpL, Tl, Tl, Kw, 1, pad);
        stats(tmpL, st0, CH * Tl);
        dwconv(xg, P(pb + 3), nullptr, tmpA, Tg, Tg, Kw, 1, pad);
        stats(tmpA, st1, CH * Tg);
        dwconv(xg, P(pb + 6), nullptr, tmpG, Tg, Tg, Kw, 1, pad);
        stats(tmpG, st2, CH * Tg);
        int tot = Bn * CH * Tl;
        la_combine_kernel<<<blocks(tot), 256, 0, stream>>>(
            tmpL, st0, P(pb + 1), P(pb + 2),
            tmpA, st1, P(pb + 4), P(pb + 5),
            tmpG, st2, P(pb + 7), P(pb + 8),
            out, CH, Tl, Tg, tot);
    };

    // ---- proj: 1x1 conv (WMMA GEMM) + GlobLN + PReLU ----
    gemm(P(I_PW), P(I_PB), P(I_X), o, CH, CIN, Tt, 0, nullptr);
    stats(o, st0, CH * Tt);
    globln(o, st0, P(I_PG), P(I_PBETA), P(I_PA), o, Tt);

    // ---- SPP pyramid (depthwise K=5, then stride-2) each + GlobLN ----
    dwconv(o, P(SPP(0, 0)), P(SPP(0, 1)), f0, Tt, Tt, 5, 1, 2);
    stats(f0, st0, CH * Tt);    globln(f0, st0, P(SPP(0, 2)), P(SPP(0, 3)), nullptr, f0, Tt);
    dwconv(f0, P(SPP(1, 0)), P(SPP(1, 1)), f1, Tt, 1024, 5, 2, 2);
    stats(f1, st0, CH * 1024);  globln(f1, st0, P(SPP(1, 2)), P(SPP(1, 3)), nullptr, f1, 1024);
    dwconv(f1, P(SPP(2, 0)), P(SPP(2, 1)), f2, 1024, 512, 5, 2, 2);
    stats(f2, st0, CH * 512);   globln(f2, st0, P(SPP(2, 2)), P(SPP(2, 3)), nullptr, f2, 512);
    dwconv(f2, P(SPP(3, 0)), P(SPP(3, 1)), f3, 512, 256, 5, 2, 2);
    stats(f3, st0, CH * 256);   globln(f3, st0, P(SPP(3, 2)), P(SPP(3, 3)), nullptr, f3, 256);

    // ---- global feature: sum of adaptive average pools ----
    { int tot = Bn * CH * Lg;
      pool_kernel<<<blocks(tot), 256, 0, stream>>>(f0, gf, Tt,   Lg, 0, tot);
      pool_kernel<<<blocks(tot), 256, 0, stream>>>(f1, gf, 1024, Lg, 1, tot);
      pool_kernel<<<blocks(tot), 256, 0, stream>>>(f2, gf, 512,  Lg, 1, tot);
      pool_kernel<<<blocks(tot), 256, 0, stream>>>(f3, gf, 256,  Lg, 1, tot); }

    // ---- S4 residual block: gf += GLU(conv_ow(gelu(SSM(LN(gf))))) ----
    lnch_kernel<<<Bn * Lg, 256, 0, stream>>>(gf, P(I_S4 + 0), P(I_S4 + 1), t1, CH, Lg);
    { int tot = CH * Lg;
      s4kern_kernel<<<blocks(tot), 256, 0, stream>>>(P(I_S4 + 2), P(I_S4 + 3),
          P(I_S4 + 4), P(I_S4 + 5), P(I_S4 + 6), kf, kb, CH, NST, Lg); }
    s4conv_kernel<<<Bn * CH, 256, 0, stream>>>(t1, kf, kb, P(I_S4 + 7), t3, CH, Lg);
    gemm(P(I_S4 + 8), P(I_S4 + 9), t3, t2, 2 * CH, CH, Lg, 0, nullptr);
    { int tot = Bn * CH * Lg;
      glu_kernel<<<blocks(tot), 256, 0, stream>>>(t2, t3, CH, Lg, tot);
      add_kernel<<<blocks(tot), 256, 0, stream>>>(gf, t3, tot); }

    // ---- FF residual block: gf += W2(gelu(W1(LN(gf)))) ----
    lnch_kernel<<<Bn * Lg, 256, 0, stream>>>(gf, P(I_FF + 0), P(I_FF + 1), t1, CH, Lg);
    gemm(P(I_FF + 2), P(I_FF + 3), t1, t2, 2 * CH, CH, Lg, 1, nullptr);
    gemm(P(I_FF + 4), P(I_FF + 5), t2, t1, CH, 2 * CH, Lg, 0, nullptr);
    { int tot = Bn * CH * Lg;
      add_kernel<<<blocks(tot), 256, 0, stream>>>(gf, t1, tot); }

    // ---- fusion LA layers (K=1) ----
    la(f0, Tt,   gf, Lg, FUSP(0), 1, fu0);
    la(f1, 1024, gf, Lg, FUSP(1), 1, fu1);
    la(f2, 512,  gf, Lg, FUSP(2), 1, fu2);
    la(f3, 256,  gf, Lg, FUSP(3), 1, fu3);   // computed (faithful), unused below

    // ---- reconstruction (faithful to reference indexing quirk) ----
    la(fu2, 512,  fu1, 1024, LSTP(2), 5, e2);
    la(fu1, 1024, e2,  512,  LSTP(1), 5, e1);
    la(fu0, Tt,   e1,  1024, LSTP(0), 5, e0);

    // ---- output 1x1 conv (WMMA) + residual add of input x ----
    gemm(P(I_RW), P(I_RB), e0, (float*)d_out, CIN, CH, Tt, 0, P(I_X));
}
